// SynoBlock_19507741458503
// MI455X (gfx1250) — compile-verified, matrix operand order checked by reading, not run on hardware
//
#include <hip/hip_runtime.h>
#include <hip/hip_bf16.h>

typedef __attribute__((ext_vector_type(16))) __bf16 v16bf;
typedef __attribute__((ext_vector_type(8)))  __bf16 v8bf;
typedef __attribute__((ext_vector_type(8)))  float  v8f;

// Problem dims (from reference): N,T,L,H,C,D,Q = 8,8,257,16,64,1024,256
constexpr int NB  = 8;
constexpr int TB  = 8;
constexpr int LB  = 257;
constexpr int HB  = 16;
constexpr int CB  = 64;
constexpr int DB  = 1024;
constexpr int QB  = 256;
constexpr int NQ  = NB * QB;     // 2048
constexpr int D4  = 4 * DB;      // 4096
constexpr int D8  = DB / 8;      // 128
constexpr int KKB = LB - 1;      // 256 (keys after dropping l=0)

// ---------------------------------------------------------------------------
// Fragment loader: 16x32 bf16 tile, row-major source with leading dim `ld`.
// Matches documented CDNA5 16-bit A layout (wave32):
//   lane<16 : row=lane,    elems 0..7 -> K k0..k0+7,  elems 8..15 -> K k0+16..k0+23
//   lane>=16: row=lane-16, elems 0..7 -> K k0+8..+15, elems 8..15 -> K k0+24..+31
// B fragments use the same pattern with N (column index) in place of M, i.e.
// B is supplied as (N x K) row-major = W, making A/B loads symmetric.
// ---------------------------------------------------------------------------
__device__ __forceinline__ v16bf load_frag(const __bf16* __restrict__ base,
                                           int ld, int row0, int k0, int lane) {
  int r  = row0 + (lane & 15);
  int kb = k0 + ((lane >> 4) << 3);
  const __bf16* p = base + (size_t)r * ld + kb;
  v8bf lo = *reinterpret_cast<const v8bf*>(p);
  v8bf hi = *reinterpret_cast<const v8bf*>(p + 16);
  v16bf f;
#pragma unroll
  for (int j = 0; j < 8; ++j) { f[j] = lo[j]; f[j + 8] = hi[j]; }
  return f;
}

#define WMMA_BF16(A_, B_, C_) \
  __builtin_amdgcn_wmma_f32_16x16x32_bf16(false, (A_), false, (B_), (short)0, (C_), false, false)

// ---------------------------------------------------------------------------
// Generic GEMM: out[M x N] = act(alpha * (A[MxK] @ W[NxK]^T + bias)) + resid
// act: 0 = none, 1 = quick_gelu. Output to bf16 (Cb) if non-null else f32 (Cf).
// One wave computes a 16(M) x 64(N) tile. blockDim = 256 (8 waves).
// ---------------------------------------------------------------------------
__global__ __launch_bounds__(256) void gemm_bf16_kernel(
    const __bf16* __restrict__ A, const __bf16* __restrict__ W,
    const float* __restrict__ bias, const float* __restrict__ resid,
    float* __restrict__ Cf, __bf16* __restrict__ Cb,
    int M, int N, int K, float alpha, int act)
{
  int lane  = threadIdx.x & 31;
  int wid   = blockIdx.x * 8 + (threadIdx.x >> 5);
  int ncols = N >> 6;
  int mt = wid / ncols;
  int nc = wid % ncols;
  if (mt >= (M >> 4)) return;

  v8f acc[4];
#pragma unroll
  for (int t = 0; t < 4; ++t)
#pragma unroll
    for (int j = 0; j < 8; ++j) acc[t][j] = 0.f;

  for (int k0 = 0; k0 < K; k0 += 32) {
    v16bf a = load_frag(A, K, mt * 16, k0, lane);
#pragma unroll
    for (int t = 0; t < 4; ++t) {
      v16bf b = load_frag(W, K, nc * 64 + t * 16, k0, lane);
      acc[t] = WMMA_BF16(a, b, acc[t]);
    }
  }

  // C/D layout: VGPR j -> row (j | j+8), lanes 0..15 / 16..31 -> cols
  int col  = lane & 15;
  int rofs = (lane >> 4) << 3;
#pragma unroll
  for (int t = 0; t < 4; ++t) {
    int n = nc * 64 + t * 16 + col;
    float bv = bias ? bias[n] : 0.f;
#pragma unroll
    for (int j = 0; j < 8; ++j) {
      int m = mt * 16 + rofs + j;
      float v = alpha * (acc[t][j] + bv);
      if (act == 1) v = v / (1.f + __expf(-1.702f * v));   // quick_gelu
      if (resid) v += resid[(size_t)m * N + n];
      if (Cb) Cb[(size_t)m * N + n] = (__bf16)v;
      else    Cf[(size_t)m * N + n] = v;
    }
  }
}

// ---------------------------------------------------------------------------
// LayerNorm over last dim, one row per block, bf16 output (GEMM A operand).
// ---------------------------------------------------------------------------
__global__ __launch_bounds__(256) void ln_rows_kernel(
    const float* __restrict__ X, const float* __restrict__ w,
    const float* __restrict__ b, __bf16* __restrict__ Y, int D)
{
  int row = blockIdx.x, tid = threadIdx.x;
  const float* x = X + (size_t)row * D;
  __shared__ float red[256];

  float s = 0.f;
  for (int i = tid; i < D; i += 256) s += x[i];
  red[tid] = s; __syncthreads();
  for (int o = 128; o > 0; o >>= 1) { if (tid < o) red[tid] += red[tid + o]; __syncthreads(); }
  float mean = red[0] / D;
  __syncthreads();

  float vs = 0.f;
  for (int i = tid; i < D; i += 256) { float d0 = x[i] - mean; vs += d0 * d0; }
  red[tid] = vs; __syncthreads();
  for (int o = 128; o > 0; o >>= 1) { if (tid < o) red[tid] += red[tid + o]; __syncthreads(); }
  float rs = rsqrtf(red[0] / D + 1e-5f);

  for (int i = tid; i < D; i += 256)
    Y[(size_t)row * D + i] = (__bf16)((x[i] - mean) * rs * w[i] + b[i]);
}

// ---------------------------------------------------------------------------
// te path (tiny, 8 rows): _te = LN(te)@l1^T@l2^T + te, scalar math is fine.
// ---------------------------------------------------------------------------
__global__ __launch_bounds__(256) void te_adaptor_kernel(
    const float* __restrict__ te, const float* __restrict__ lnw,
    const float* __restrict__ lnb, const float* __restrict__ l1,
    const float* __restrict__ l2, float* __restrict__ te_hat)
{
  int row = blockIdx.x, tid = threadIdx.x;
  __shared__ float xr[DB];
  __shared__ float hh[D8];
  __shared__ float red[256];
  const float* x = te + (size_t)row * DB;

  float s = 0.f;
  for (int i = tid; i < DB; i += 256) s += x[i];
  red[tid] = s; __syncthreads();
  for (int o = 128; o > 0; o >>= 1) { if (tid < o) red[tid] += red[tid + o]; __syncthreads(); }
  float mean = red[0] / DB;
  __syncthreads();
  float vs = 0.f;
  for (int i = tid; i < DB; i += 256) { float d0 = x[i] - mean; vs += d0 * d0; }
  red[tid] = vs; __syncthreads();
  for (int o = 128; o > 0; o >>= 1) { if (tid < o) red[tid] += red[tid + o]; __syncthreads(); }
  float rs = rsqrtf(red[0] / DB + 1e-5f);
  for (int i = tid; i < DB; i += 256) xr[i] = (x[i] - mean) * rs * lnw[i] + lnb[i];
  __syncthreads();

  if (tid < D8) {
    float a = 0.f;
    for (int k = 0; k < DB; ++k) a += xr[k] * l1[(size_t)tid * DB + k];
    hh[tid] = a;
  }
  __syncthreads();

  for (int o = tid; o < DB; o += 256) {
    float a = 0.f;
    for (int k = 0; k < D8; ++k) a += hh[k] * l2[(size_t)o * D8 + k];
    te_hat[(size_t)row * DB + o] = a + x[o];
  }
}

// te_qkv: only K and V thirds are needed. grid (T, 8), 256 threads.
__global__ __launch_bounds__(256) void te_kv_kernel(
    const float* __restrict__ te_hat, const float* __restrict__ W,
    const float* __restrict__ bias, float* __restrict__ tek, float* __restrict__ tev)
{
  int t = blockIdx.x;
  int o = blockIdx.y * 256 + threadIdx.x;          // [0, 2*D)
  int cidx = DB + o;                                // cols [D, 3D)
  const float* wrow = W + (size_t)cidx * DB;
  const float* xr   = te_hat + (size_t)t * DB;
  float a = bias[cidx];
  for (int k = 0; k < DB; ++k) a += xr[k] * wrow[k];
  if (o < DB) tek[(size_t)t * DB + o]      = a;
  else        tev[(size_t)t * DB + o - DB] = a;
}

// ---------------------------------------------------------------------------
// Pack K/V: keff(n,t,k,h,c) = bf16(attrs_k[n,t,1+k,h,c] + te_k[t,h,c])
//           vT  (n,t,h,c,k) = bf16(attrs_v[n,t,1+k,h,c] + te_v[t,h,c])
// ---------------------------------------------------------------------------
__global__ __launch_bounds__(256) void pack_kv_kernel(
    const float* __restrict__ ak, const float* __restrict__ av,
    const float* __restrict__ tek, const float* __restrict__ tev,
    __bf16* __restrict__ keff, __bf16* __restrict__ vT)
{
  size_t total = (size_t)NB * TB * KKB * HB * CB;
  for (size_t idx = (size_t)blockIdx.x * 256 + threadIdx.x; idx < total;
       idx += (size_t)gridDim.x * 256) {
    int c = (int)(idx % CB); size_t r = idx / CB;
    int h = (int)(r % HB);   r /= HB;
    int k = (int)(r % KKB);  r /= KKB;
    int t = (int)(r % TB);
    int n = (int)(r / TB);
    size_t src = ((((size_t)n * TB + t) * LB + (1 + k)) * HB + h) * CB + c;
    int dch = h * CB + c;
    keff[idx] = (__bf16)(ak[src] + tek[(size_t)t * DB + dch]);
    vT[(((size_t)(n * TB + t) * HB + h) * CB + c) * KKB + k] =
        (__bf16)(av[src] + tev[(size_t)t * DB + dch]);
  }
}

// ---------------------------------------------------------------------------
// Fused attention per (n,t,h): scores = sq(Q x C) @ keff^T + mask, softmax(k),
// mix = probs @ veff. 8 waves x 2 q-tiles of 16 rows. All WMMA.
// ---------------------------------------------------------------------------
__global__ __launch_bounds__(256) void attn_kernel(
    const __bf16* __restrict__ sq,    // (N,Q,H,C), already *1/sqrt(C) + bias
    const __bf16* __restrict__ keff,  // (N,T,KK,H,C)
    const __bf16* __restrict__ vT,    // (N,T,H,C,KK)
    const float*  __restrict__ pmask, // (Q, KK)
    float* __restrict__ smix)         // (N,T,Q,D)
{
  __shared__ __align__(16) __bf16 pbuf[8][16][32];  // per-wave prob staging (8 KB)

  int blk = blockIdx.x;                  // n*T*H + t*H + h
  int h  = blk % HB;
  int nt = blk / HB;
  int t  = nt % TB;
  int n  = nt / TB;
  int tid = threadIdx.x, wave = tid >> 5, lane = tid & 31;
  int col = lane & 15, rofs = (lane >> 4) << 3;

  const __bf16* qbase = sq   + ((size_t)n * QB * HB + h) * CB;          // row=q, ld=H*C
  const __bf16* kbase = keff + ((size_t)(n * TB + t) * KKB * HB + h) * CB; // row=k, ld=H*C
  const __bf16* vbase = vT   + ((size_t)(n * TB + t) * HB + h) * (size_t)CB * KKB; // row=c, ld=KK

  for (int qi = 0; qi < 2; ++qi) {
    int q0 = (wave * 2 + qi) * 16;
    v16bf aq0 = load_frag(qbase, HB * CB, q0, 0,  lane);
    v16bf aq1 = load_frag(qbase, HB * CB, q0, 32, lane);

    // scores: 16 tiles of 16x16 over k, K-dim = C = 64 (two wmma each)
    v8f acc[16];
#pragma unroll
    for (int kt = 0; kt < 16; ++kt)
#pragma unroll
      for (int j = 0; j < 8; ++j) acc[kt][j] = 0.f;

#pragma unroll 4
    for (int kt = 0; kt < 16; ++kt) {
      v16bf b0 = load_frag(kbase, HB * CB, kt * 16, 0,  lane);
      acc[kt] = WMMA_BF16(aq0, b0, acc[kt]);
      v16bf b1 = load_frag(kbase, HB * CB, kt * 16, 32, lane);
      acc[kt] = WMMA_BF16(aq1, b1, acc[kt]);
    }

    // mask + row softmax (row = q, spread across 16-lane half + register kt)
    float rmax[8];
#pragma unroll
    for (int j = 0; j < 8; ++j) rmax[j] = -3.4e38f;
    for (int kt = 0; kt < 16; ++kt)
#pragma unroll
      for (int j = 0; j < 8; ++j) {
        float v = acc[kt][j] + pmask[(size_t)(q0 + rofs + j) * KKB + kt * 16 + col];
        acc[kt][j] = v;
        rmax[j] = fmaxf(rmax[j], v);
      }
#pragma unroll
    for (int j = 0; j < 8; ++j)
      for (int o = 1; o < 16; o <<= 1)
        rmax[j] = fmaxf(rmax[j], __shfl_xor(rmax[j], o, 16));

    float rsum[8];
#pragma unroll
    for (int j = 0; j < 8; ++j) rsum[j] = 0.f;
    for (int kt = 0; kt < 16; ++kt)
#pragma unroll
      for (int j = 0; j < 8; ++j) {
        float e = __expf(acc[kt][j] - rmax[j]);
        acc[kt][j] = e;
        rsum[j] += e;
      }
#pragma unroll
    for (int j = 0; j < 8; ++j)
      for (int o = 1; o < 16; o <<= 1)
        rsum[j] += __shfl_xor(rsum[j], o, 16);
    float rinv[8];
#pragma unroll
    for (int j = 0; j < 8; ++j) rinv[j] = 1.f / rsum[j];

    // mix: probs(16 x 256) @ veff(256 x 64), probs repacked via LDS to A-layout
    v8f om[4];
#pragma unroll
    for (int ct = 0; ct < 4; ++ct)
#pragma unroll
      for (int j = 0; j < 8; ++j) om[ct][j] = 0.f;

    for (int kc = 0; kc < 8; ++kc) {
#pragma unroll
      for (int half = 0; half < 2; ++half) {
        int kt = kc * 2 + half;
#pragma unroll
        for (int j = 0; j < 8; ++j)
          pbuf[wave][rofs + j][half * 16 + col] = (__bf16)(acc[kt][j] * rinv[j]);
      }
      v16bf pa = load_frag(&pbuf[wave][0][0], 32, 0, 0, lane);
#pragma unroll
      for (int ct = 0; ct < 4; ++ct) {
        v16bf vb = load_frag(vbase, KKB, ct * 16, kc * 32, lane);
        om[ct] = WMMA_BF16(pa, vb, om[ct]);
      }
    }

#pragma unroll
    for (int ct = 0; ct < 4; ++ct)
#pragma unroll
      for (int j = 0; j < 8; ++j)
        smix[((size_t)(n * TB + t) * QB + q0 + rofs + j) * DB + h * CB + ct * 16 + col] =
            om[ct][j];
  }
}

// ---------------------------------------------------------------------------
// Depthwise temporal conv (k=3, pad 1) + bias + residual, mean over T -> bf16.
// smix layout (n,t,q,d); xm[nq,d,t] maps to stride Q*D in t.
// ---------------------------------------------------------------------------
__global__ __launch_bounds__(256) void conv_mean_kernel(
    const float* __restrict__ smix, const float* __restrict__ tw,
    const float* __restrict__ tb, __bf16* __restrict__ meanb)
{
  size_t total = (size_t)NQ * DB;
  for (size_t idx = (size_t)blockIdx.x * 256 + threadIdx.x; idx < total;
       idx += (size_t)gridDim.x * 256) {
    int d = (int)(idx % DB);
    size_t nq = idx / DB;
    int n = (int)(nq / QB), q = (int)(nq % QB);
    float xv[TB];
#pragma unroll
    for (int t = 0; t < TB; ++t)
      xv[t] = smix[((size_t)(n * TB + t) * QB + q) * DB + d];
    float w0 = tw[d * 3], w1 = tw[d * 3 + 1], w2 = tw[d * 3 + 2], bb = tb[d];
    float acc = 0.f;
#pragma unroll
    for (int t = 0; t < TB; ++t) {
      float y = w1 * xv[t];
      if (t > 0)      y += w0 * xv[t - 1];
      if (t < TB - 1) y += w2 * xv[t + 1];
      acc += y + bb + xv[t];
    }
    meanb[idx] = (__bf16)(acc * (1.f / TB));
  }
}

// f32 -> bf16 conversion (weights / operands).
__global__ __launch_bounds__(256) void cvt_bf16_kernel(
    const float* __restrict__ s, __bf16* __restrict__ d, size_t n)
{
  for (size_t i = (size_t)blockIdx.x * 256 + threadIdx.x; i < n;
       i += (size_t)gridDim.x * 256)
    d[i] = (__bf16)s[i];
}

// ---------------------------------------------------------------------------
extern "C" void kernel_launch(void* const* d_in, const int* in_sizes, int n_in,
                              void* d_out, int out_size, void* d_ws, size_t ws_size,
                              hipStream_t stream)
{
  (void)in_sizes; (void)n_in; (void)out_size; (void)ws_size;
  const float* attrs_k = (const float*)d_in[0];
  const float* attrs_v = (const float*)d_in[1];
  const float* s_in    = (const float*)d_in[2];
  const float* te      = (const float*)d_in[3];
  const float* pmask   = (const float*)d_in[4];
  const float* ln1_w   = (const float*)d_in[5];
  const float* ln1_b   = (const float*)d_in[6];
  const float* in_w    = (const float*)d_in[7];
  const float* in_b    = (const float*)d_in[8];
  const float* out_w   = (const float*)d_in[9];
  const float* out_b   = (const float*)d_in[10];
  const float* ln2_w   = (const float*)d_in[11];
  const float* ln2_b   = (const float*)d_in[12];
  const float* fc_w    = (const float*)d_in[13];
  const float* fc_b    = (const float*)d_in[14];
  const float* pj_w    = (const float*)d_in[15];
  const float* pj_b    = (const float*)d_in[16];
  const float* syln_w  = (const float*)d_in[17];
  const float* syln_b  = (const float*)d_in[18];
  const float* sy1_w   = (const float*)d_in[19];
  const float* sy2_w   = (const float*)d_in[20];
  const float* teln_w  = (const float*)d_in[21];
  const float* teln_b  = (const float*)d_in[22];
  const float* te1_w   = (const float*)d_in[23];
  const float* te2_w   = (const float*)d_in[24];
  const float* tc_w    = (const float*)d_in[25];
  const float* tc_b    = (const float*)d_in[26];

  char* ws = (char*)d_ws;
  size_t off = 0;
  auto alloc = [&](size_t bytes) -> void* {
    void* p = ws + off;
    off = (off + bytes + 255) & ~(size_t)255;
    return p;
  };

  __bf16* w_inq  = (__bf16*)alloc((size_t)DB * DB * 2);   // in_proj rows [0,D)
  __bf16* w_out  = (__bf16*)alloc((size_t)DB * DB * 2);
  __bf16* w_fc   = (__bf16*)alloc((size_t)D4 * DB * 2);
  __bf16* w_pj   = (__bf16*)alloc((size_t)DB * D4 * 2);
  __bf16* w_sy1  = (__bf16*)alloc((size_t)D8 * DB * 2);
  __bf16* w_sy2  = (__bf16*)alloc((size_t)DB * D8 * 2);
  __bf16* ln_bf  = (__bf16*)alloc((size_t)NQ * DB * 2);   // reused for LN outputs
  __bf16* h1_bf  = (__bf16*)alloc((size_t)NQ * D8 * 2);
  float*  s_hat  = (float*) alloc((size_t)NQ * DB * 4);   // _s
  __bf16* sq_bf  = (__bf16*)alloc((size_t)NQ * DB * 2);   // scaled queries
  float*  te_hat = (float*) alloc((size_t)TB * DB * 4);
  float*  tek    = (float*) alloc((size_t)TB * DB * 4);
  float*  tev    = (float*) alloc((size_t)TB * DB * 4);
  __bf16* keff   = (__bf16*)alloc((size_t)NB * TB * KKB * HB * CB * 2);
  __bf16* vT     = (__bf16*)alloc((size_t)NB * TB * HB * CB * KKB * 2);
  float*  smix   = (float*) alloc((size_t)NB * TB * QB * DB * 4);
  __bf16* mean_b = (__bf16*)alloc((size_t)NQ * DB * 2);
  float*  s1     = (float*) alloc((size_t)NQ * DB * 4);
  __bf16* hmid   = (__bf16*)alloc((size_t)NQ * D4 * 2);

  auto cvt = [&](const float* src, __bf16* dst, size_t n) {
    int blocks = (int)((n + 255) / 256);
    if (blocks > 8192) blocks = 8192;
    cvt_bf16_kernel<<<blocks, 256, 0, stream>>>(src, dst, n);
  };
  auto gemm = [&](const __bf16* A, const __bf16* W, const float* bias,
                  const float* resid, float* Cf, __bf16* Cb,
                  int M, int N, int K, float alpha, int act) {
    int waves  = (M / 16) * (N / 64);
    int blocks = (waves + 7) / 8;
    gemm_bf16_kernel<<<blocks, 256, 0, stream>>>(A, W, bias, resid, Cf, Cb,
                                                 M, N, K, alpha, act);
  };

  // ---- weight conversions (bf16 GEMM B operands) ----
  cvt(in_w,  w_inq, (size_t)DB * DB);   // first D rows of in_proj_w
  cvt(out_w, w_out, (size_t)DB * DB);
  cvt(fc_w,  w_fc,  (size_t)D4 * DB);
  cvt(pj_w,  w_pj,  (size_t)DB * D4);
  cvt(sy1_w, w_sy1, (size_t)D8 * DB);
  cvt(sy2_w, w_sy2, (size_t)DB * D8);

  // ---- syno adaptor: _s = s + LN(s)@l1^T@l2^T ----
  ln_rows_kernel<<<NQ, 256, 0, stream>>>(s_in, syln_w, syln_b, ln_bf, DB);
  gemm(ln_bf, w_sy1, nullptr, nullptr, nullptr, h1_bf, NQ, D8, DB, 1.f, 0);
  gemm(h1_bf, w_sy2, nullptr, s_in, s_hat, nullptr, NQ, DB, D8, 1.f, 0);

  // ---- queries: sq = (LN(_s)@Wq^T + bq) * 1/sqrt(C) ----
  ln_rows_kernel<<<NQ, 256, 0, stream>>>(s_hat, ln1_w, ln1_b, ln_bf, DB);
  gemm(ln_bf, w_inq, in_b, nullptr, nullptr, sq_bf, NQ, DB, DB, 0.125f, 0);

  // ---- te path: _te then te_k/te_v ----
  te_adaptor_kernel<<<TB, 256, 0, stream>>>(te, teln_w, teln_b, te1_w, te2_w, te_hat);
  te_kv_kernel<<<dim3(TB, 8), 256, 0, stream>>>(te_hat, in_w, in_b, tek, tev);

  // ---- pack keff / vT (adds te_k, te_v; drops l=0) ----
  pack_kv_kernel<<<8192, 256, 0, stream>>>(attrs_k, attrs_v, tek, tev, keff, vT);

  // ---- fused WMMA attention ----
  attn_kernel<<<NB * TB * HB, 256, 0, stream>>>(sq_bf, keff, vT, pmask, smix);

  // ---- temporal conv + residual + mean over T ----
  conv_mean_kernel<<<8192, 256, 0, stream>>>(smix, tc_w, tc_b, mean_b);

  // ---- out_proj + residual(s): s1 ----
  gemm(mean_b, w_out, out_b, s_in, s1, nullptr, NQ, DB, DB, 1.f, 0);

  // ---- MLP: s1 + c_proj(quick_gelu(c_fc(LN(s1)))) ----
  ln_rows_kernel<<<NQ, 256, 0, stream>>>(s1, ln2_w, ln2_b, ln_bf, DB);
  gemm(ln_bf, w_fc, fc_b, nullptr, nullptr, hmid, NQ, D4, DB, 1.f, 1);
  gemm(hmid, w_pj, pj_b, s1, (float*)d_out, nullptr, NQ, DB, D4, 1.f, 0);
}
